// GCNEncoder_66838281060450
// MI455X (gfx1250) — matrix operand order repaired and checked by reading, not compiled, not run on hardware
//
#include <hip/hip_runtime.h>
#include <cstdint>
#include <cstddef>

// ---------------------------------------------------------------------------
// GAT (6 layers) for MI455X / gfx1250, wave32.
//   - Dense transforms X@W via V_WMMA_F32_16X16X32_F16 (f32 accumulate)
//   - Edge softmax/aggregation via f32 global atomics (memory-bound part)
// ---------------------------------------------------------------------------

typedef __attribute__((ext_vector_type(16))) _Float16 v16h;
typedef __attribute__((ext_vector_type(8)))  float    v8f;

#define ENC_NEG_INF 0x007FFFFFu   // fenc(-inf)

__device__ __forceinline__ unsigned fenc(float f) {
  unsigned u = __float_as_uint(f);
  return (u & 0x80000000u) ? ~u : (u | 0x80000000u);
}
__device__ __forceinline__ float fdec(unsigned e) {
  unsigned b = (e & 0x80000000u) ? (e & 0x7FFFFFFFu) : ~e;
  return __uint_as_float(b);
}

// ---- convert activation (f32, N x K) -> zero-padded f16 (Npad x Kp) --------
__global__ void k_cvt_x(const float* __restrict__ x, _Float16* __restrict__ xh,
                        int N, int K, int Npad, int Kp) {
  int i = blockIdx.x * blockDim.x + threadIdx.x;
  long long total = (long long)Npad * Kp;
  if (i >= total) return;
  int n = i / Kp, k = i - n * Kp;
  float v = (n < N && k < K) ? x[(size_t)n * K + k] : 0.0f;
  xh[i] = (_Float16)v;
}

// ---- convert weight (f32, K x M) -> transposed padded f16 (Mp x Kp) --------
__global__ void k_cvt_wt(const float* __restrict__ w, _Float16* __restrict__ wt,
                         int K, int M, int Kp, int Mp) {
  int i = blockIdx.x * blockDim.x + threadIdx.x;
  if (i >= Mp * Kp) return;
  int m = i / Kp, k = i - m * Kp;
  float v = (m < M && k < K) ? w[(size_t)k * M + m] : 0.0f;
  wt[i] = (_Float16)v;
}

// ---- WMMA GEMM: out[N x M] = Xh[Npad x Kp] * Wt^T (Wt is Mp x Kp) ----------
// One wave computes one 16x16 tile; 4 waves/block cover 4 M-tiles.
__global__ __launch_bounds__(128)
void k_wmma_gemm(const _Float16* __restrict__ Xh, const _Float16* __restrict__ Wt,
                 float* __restrict__ out, int N, int M, int Kp, int mtiles) {
  const int lane  = threadIdx.x & 31;
  const int wave  = threadIdx.x >> 5;
  const int ntile = blockIdx.x;
  const int mtile = blockIdx.y * 4 + wave;
  if (mtile >= mtiles) return;                    // wave-uniform

  const bool hi    = lane >= 16;
  const int  hiOff = hi ? 8 : 0;
  const int  row   = ntile * 16 + (lane & 15);    // A row (node)
  const int  col   = mtile * 16 + (lane & 15);    // B/C column

  v8f acc = {};
  const uint32_t* arow = (const uint32_t*)(Xh + (size_t)row * Kp);
  const uint32_t* brow = (const uint32_t*)(Wt + (size_t)col * Kp);

  for (int k0 = 0; k0 < Kp; k0 += 32) {
    union { v16h v; uint32_t u[8]; } a, b;
    const uint32_t* ap = arow + (k0 >> 1);
    const uint32_t* bp = brow + (k0 >> 1);
#pragma unroll
    for (int v = 0; v < 8; ++v) {
      // A 16x32 f16 layout: VGPR v<4 -> K = hiOff+2v ; v>=4 -> K = 16+hiOff+2(v-4)
      int kA = (v < 4) ? (hiOff + 2 * v) : (16 + hiOff + 2 * (v - 4));
      a.u[v] = ap[kA >> 1];
      // B 32x16 f16 layout: same K order as A: v<4 -> K = hiOff+2v ; v>=4 -> K = 16+hiOff+2(v-4)
      int kB = (v < 4) ? (hiOff + 2 * v) : (16 + hiOff + 2 * (v - 4));
      b.u[v] = bp[kB >> 1];
    }
    acc = __builtin_amdgcn_wmma_f32_16x16x32_f16(
        /*neg_a=*/false, a.v, /*neg_b=*/false, b.v,
        /*c_mod=*/(short)0, acc, /*reuse_a=*/false, /*reuse_b=*/false);
  }

  // C/D 16x16 f32: VGPR v -> row = v + (lane>=16 ? 8 : 0), col = lane&15
  const int rbase = ntile * 16 + hiOff;
#pragma unroll
  for (int v = 0; v < 8; ++v) {
    int r = rbase + v;
    if (r < N && col < M) out[(size_t)r * M + col] = acc[v];
  }
}

// ---- per-node attention dots: es/ed[n,h] = sum_c xw[n,h,c]*a{s,d}[h,c] -----
__global__ void k_attn_dots(const float* __restrict__ xw,
                            const float* __restrict__ as_, const float* __restrict__ ad_,
                            float* __restrict__ es, float* __restrict__ ed,
                            int N, int H, int C) {
  int i = blockIdx.x * blockDim.x + threadIdx.x;
  if (i >= N * H) return;
  int h = i % H;
  const float* v = xw + (size_t)i * C;   // i = n*H + h, xw row-major [N, H*C]
  float s = 0.f, d = 0.f;
  for (int c = 0; c < C; ++c) { s += v[c] * as_[h * C + c]; d += v[c] * ad_[h * C + c]; }
  es[i] = s; ed[i] = d;
}

// ---- init accumulators -----------------------------------------------------
__global__ void k_zero(float* __restrict__ p, long long n) {
  long long i = (long long)blockIdx.x * blockDim.x + threadIdx.x;
  if (i < n) p[i] = 0.0f;
}
__global__ void k_init_softmax(float* __restrict__ denom, unsigned* __restrict__ emax, int n) {
  int i = blockIdx.x * blockDim.x + threadIdx.x;
  if (i < n) { denom[i] = 0.0f; emax[i] = ENC_NEG_INF; }
}

// ---- edge pass 1: e = leakyrelu(es[src]+ed[dst]); segment-max by dst -------
__global__ void k_edge_max(const int* __restrict__ ei, int E, int Etot,
                           const float* __restrict__ es, const float* __restrict__ ed,
                           float* __restrict__ eval, unsigned* __restrict__ emax, int H) {
  int i = blockIdx.x * blockDim.x + threadIdx.x;
  if (i >= Etot * H) return;
  int e = i / H, h = i - e * H;
  int src = (e < E) ? ei[e] : (e - E);
  int dst = (e < E) ? ei[E + e] : (e - E);
  float v = es[src * H + h] + ed[dst * H + h];
  v = (v > 0.f) ? v : 0.2f * v;                     // leaky_relu(0.2)
  eval[i] = v;
  atomicMax(&emax[dst * H + h], fenc(v));
}

// ---- edge pass 2: ee = exp(e - emax[dst]); segment-sum by dst --------------
__global__ void k_edge_expsum(const int* __restrict__ ei, int E, int Etot,
                              float* __restrict__ eval, const unsigned* __restrict__ emax,
                              float* __restrict__ denom, int H) {
  int i = blockIdx.x * blockDim.x + threadIdx.x;
  if (i >= Etot * H) return;
  int e = i / H, h = i - e * H;
  int dst = (e < E) ? ei[E + e] : (e - E);
  float ee = __expf(eval[i] - fdec(emax[dst * H + h]));
  eval[i] = ee;
  atomicAdd(&denom[dst * H + h], ee);
}

// ---- edge pass 3: agg[dst] += xw[src] * alpha (4 channels / thread) --------
__global__ void k_edge_agg(const int* __restrict__ ei, int E, int Etot,
                           const float* __restrict__ eval, const float* __restrict__ denom,
                           const float* __restrict__ xw, float* __restrict__ agg,
                           int H, int C, int C4) {
  long long i = (long long)blockIdx.x * blockDim.x + threadIdx.x;
  if (i >= (long long)Etot * H * C4) return;
  int c4 = (int)(i % C4);
  long long t = i / C4;
  int h = (int)(t % H);
  int e = (int)(t / H);
  int src = (e < E) ? ei[e] : (e - E);
  int dst = (e < E) ? ei[E + e] : (e - E);
  float alpha = eval[(size_t)e * H + h] / denom[dst * H + h];
  int M = H * C;
  const float* xs = xw + (size_t)src * M + h * C;
  float* ag = agg + (size_t)dst * M + h * C;
#pragma unroll
  for (int cc = 0; cc < 4; ++cc) {
    int c = c4 * 4 + cc;
    if (c < C) atomicAdd(&ag[c], xs[c] * alpha);
  }
}

// ---- finalize: + bias, optional relu ---------------------------------------
__global__ void k_finalize(const float* __restrict__ agg, const float* __restrict__ bias,
                           float* __restrict__ dst, long long total, int M, int do_relu) {
  long long i = (long long)blockIdx.x * blockDim.x + threadIdx.x;
  if (i >= total) return;
  float v = agg[i] + bias[(int)(i % M)];
  if (do_relu) v = v > 0.f ? v : 0.f;
  dst[i] = v;
}

// ---------------------------------------------------------------------------
static inline int cdiv(long long a, int b) { return (int)((a + b - 1) / b); }

extern "C" void kernel_launch(void* const* d_in, const int* in_sizes, int n_in,
                              void* d_out, int out_size, void* d_ws, size_t ws_size,
                              hipStream_t stream) {
  const float* x  = (const float*)d_in[0];
  const int*   ei = (const int*)d_in[1];
  const int N = in_sizes[0] / 20;
  const int E = in_sizes[1] / 2;
  const int Etot = E + N;              // PyG self-loops appended
  const int Npad = ((N + 15) / 16) * 16;

  // layer specs: fin, H, C (M = H*C); last layer has concat=False but H==1
  const int LF[6] = {20, 320, 100, 80, 40, 20};
  const int LH[6] = {8, 1, 1, 1, 1, 1};
  const int LC[6] = {40, 100, 80, 40, 20, 3};

  // ---- workspace carving (256B aligned slices) ----
  size_t off = 0;
  auto carve = [&](size_t bytes) {
    void* p = (char*)d_ws + off;
    off += (bytes + 255) & ~(size_t)255;
    return p;
  };
  const int MAXM = 320, MAXH = 8, MAXKP = 320;
  float*     bufA  = (float*)carve((size_t)N * MAXM * 4);
  float*     bufB  = (float*)carve((size_t)N * MAXM * 4);
  float*     xw    = (float*)carve((size_t)N * MAXM * 4);
  _Float16*  Xh    = (_Float16*)carve((size_t)Npad * MAXKP * 2);
  _Float16*  Wth   = (_Float16*)carve((size_t)MAXM * MAXKP * 2);   // padded Mp x Kp
  float*     es    = (float*)carve((size_t)N * MAXH * 4);
  float*     ed    = (float*)carve((size_t)N * MAXH * 4);
  float*     denom = (float*)carve((size_t)N * MAXH * 4);
  unsigned*  emax  = (unsigned*)carve((size_t)N * MAXH * 4);
  float*     evalb = (float*)carve((size_t)Etot * MAXH * 4);
  (void)ws_size; (void)n_in;

  const int B = 256;
  const float* cur = x;
  float* aggbufs[2] = {bufA, bufB};

  for (int L = 0; L < 6; ++L) {
    const int K = LF[L], H = LH[L], C = LC[L], M = H * C;
    const int Kp = ((K + 31) / 32) * 32;
    const int Mp = ((M + 15) / 16) * 16;
    const int mtiles = Mp / 16;
    const float* W  = (const float*)d_in[2 + 4 * L];
    const float* as_ = (const float*)d_in[3 + 4 * L];
    const float* ad_ = (const float*)d_in[4 + 4 * L];
    const float* bb = (const float*)d_in[5 + 4 * L];
    float* agg = aggbufs[L & 1];

    // 1) f16 staging (zero-padded)
    k_cvt_x<<<cdiv((long long)Npad * Kp, B), B, 0, stream>>>(cur, Xh, N, K, Npad, Kp);
    k_cvt_wt<<<cdiv((long long)Mp * Kp, B), B, 0, stream>>>(W, Wth, K, M, Kp, Mp);

    // 2) XW via WMMA
    dim3 gg(Npad / 16, (mtiles + 3) / 4);
    k_wmma_gemm<<<gg, 128, 0, stream>>>(Xh, Wth, xw, N, M, Kp, mtiles);

    // 3) attention dots
    k_attn_dots<<<cdiv((long long)N * H, B), B, 0, stream>>>(xw, as_, ad_, es, ed, N, H, C);

    // 4) init accumulators
    k_zero<<<cdiv((long long)N * M, B), B, 0, stream>>>(agg, (long long)N * M);
    k_init_softmax<<<cdiv((long long)N * H, B), B, 0, stream>>>(denom, emax, N * H);

    // 5) edge-wise softmax + aggregation
    k_edge_max<<<cdiv((long long)Etot * H, B), B, 0, stream>>>(ei, E, Etot, es, ed, evalb, emax, H);
    k_edge_expsum<<<cdiv((long long)Etot * H, B), B, 0, stream>>>(ei, E, Etot, evalb, emax, denom, H);
    const int C4 = (C + 3) / 4;
    k_edge_agg<<<cdiv((long long)Etot * H * C4, B), B, 0, stream>>>(ei, E, Etot, evalb, denom, xw, agg, H, C, C4);

    // 6) bias (+relu), write next activation or final output
    float* dst = (L == 5) ? (float*)d_out : agg;
    k_finalize<<<cdiv((long long)N * M, B), B, 0, stream>>>(agg, bb, dst, (long long)N * M, M, L < 5 ? 1 : 0);
    cur = agg;
  }
  (void)out_size;
}
